// ProteinLigandInteractionNetwork_79405355369231
// MI455X (gfx1250) — matrix operand order repaired and checked by reading, not compiled, not run on hardware
//
#include <hip/hip_runtime.h>
#include <hip/hip_bf16.h>
#include <math.h>

// ---------------------------------------------------------------------------
// MI455X (gfx1250) implementation, wave32 WMMA (v_wmma_f32_16x16x32_f16).
// - Fused pair pipeline: geometric feats -> MLP1(silu) -> MLP2 -> head bias,
//   all in LDS; only the mandatory 240MB pair tensor + 60MB f16 bias hit HBM.
// - All LDS fragment reads are 16B ds_load_b128 (A row-major, B staged
//   N-major so per-lane K-runs are contiguous).
// - All global staging is float4 (global_load_b128), branchless M-tail.
// - Fast-path activations via v_rcp_f32 (__builtin_amdgcn_rcpf).
// Requires K % 32 == 0 at every GEMM call site (true: 64/224/256/512).
// ---------------------------------------------------------------------------

typedef __attribute__((ext_vector_type(16))) _Float16 v16h;
typedef __attribute__((ext_vector_type(8)))  _Float16 v8h;
typedef __attribute__((ext_vector_type(4)))  _Float16 v4h;
typedef __attribute__((ext_vector_type(8)))  float    v8f;

#define BATCH 4
#define NSEQ 484
#define NPROT 420
#define NPAD 512
#define HID 256
#define HEADS 8
#define DH 32
#define PAIR_M (BATCH * NSEQ * NSEQ)   // 937024, multiple of 64

// ---- WMMA fragment builders (ISA 7.12.2 layouts, wave32) -------------------
// A tile in LDS: [16 rows][ld cols] f16, row stride `ld` (multiple of 8).
// lanes 0-15: row=lane, elems 0..7 = K0..7, 8..15 = K16..23
// lanes 16-31: row=lane-16, elems 0..7 = K8..15, 8..15 = K24..31
// -> two contiguous 16B ds_load_b128 per lane.
__device__ inline v16h frag_a(const _Float16* s, int ld, int lane) {
  int row = lane & 15;
  int kh = (lane >> 4) * 8;
  const _Float16* p = s + row * ld + kh;
  v8h lo = *(const v8h*)p;
  v8h hi = *(const v8h*)(p + 16);
  v16h a;
#pragma unroll
  for (int e = 0; e < 8; ++e) { a[e] = lo[e]; a[e + 8] = hi[e]; }
  return a;
}
// B tile staged N-major in LDS: [16 cols][ldc k-elems], col stride `ldc`.
// lanes 0-15: col=lane, K 0..7,16..23 ; lanes 16-31: col=lane-16, K 8..15,24..31
// -> two contiguous 16B ds_load_b128 per lane.
__device__ inline v16h frag_bT(const _Float16* s, int ldc, int lane) {
  int col = lane & 15;
  int kb = (lane >> 4) * 8;
  const _Float16* p = s + col * ldc + kb;
  v8h lo = *(const v8h*)p;
  v8h hi = *(const v8h*)(p + 16);
  v16h b;
#pragma unroll
  for (int e = 0; e < 8; ++e) { b[e] = lo[e]; b[e + 8] = hi[e]; }
  return b;
}

__device__ inline v8f wmma_f16(v16h a, v16h b, v8f c) {
  return __builtin_amdgcn_wmma_f32_16x16x32_f16(false, a, false, b, (short)0, c,
                                                false, false);
}

// silu via v_exp_f32 + v_rcp_f32 (no precise-division sequence)
__device__ inline float silu(float x) {
  return x * __builtin_amdgcn_rcpf(1.f + __expf(-x));
}

// Copy a 16x32 f32 tile (row stride gs) to LDS [16][32] f16. float4 loads,
// v4h (8B) stores; no bounds checks (callers guarantee in-range).
__device__ inline void stage_tile(const float* __restrict__ G, size_t gs,
                                  _Float16* S, int tid, int nthreads) {
  for (int t = tid; t < 128; t += nthreads) {
    int r = t >> 3, c4 = (t & 7) * 4;
    float4 f = *(const float4*)(G + (size_t)r * gs + c4);
    v4h h4 = {(_Float16)f.x, (_Float16)f.y, (_Float16)f.z, (_Float16)f.w};
    *(v4h*)(S + r * 32 + c4) = h4;
  }
}
// Copy a 32x16 f32 tile (k rows, stride gs; 16 contiguous cols) to LDS
// N-major [16 cols][32 k] f16. float4 (coalesced) loads, b16 LDS scatter.
__device__ inline void stage_bT(const float* __restrict__ G, size_t gs,
                                _Float16* S, int lane) {
  for (int t = lane; t < 128; t += 32) {
    int kr = t >> 2, c4 = (t & 3) * 4;
    float4 f = *(const float4*)(G + (size_t)kr * gs + c4);
    S[(c4 + 0) * 32 + kr] = (_Float16)f.x;
    S[(c4 + 1) * 32 + kr] = (_Float16)f.y;
    S[(c4 + 2) * 32 + kr] = (_Float16)f.z;
    S[(c4 + 3) * 32 + kr] = (_Float16)f.w;
  }
}

// ---------------------------------------------------------------------------
// Generic GEMM: C[M,Nc] = act(A_f32[M,K] @ W_f32[K,Nc] + bias)
// block = 128 threads = 4 waves; wave w owns N-tile blockIdx.y*64 + w*16.
// A tile shared by the 4 waves; per-wave B tile; K stepped by 32.
// ACT resolved at compile time; bias is always non-null.
// ---------------------------------------------------------------------------
template <int ACT>
__global__ void gemm_bias_act(const float* __restrict__ A,
                              const float* __restrict__ W,
                              const float* __restrict__ bias,
                              float* __restrict__ C,
                              int M, int K, int Nc) {
  __shared__ __align__(16) _Float16 sA[16 * 32];
  __shared__ __align__(16) _Float16 sB[4][16 * 32];
  int tid = threadIdx.x, lane = tid & 31, w = tid >> 5;
  int m0 = blockIdx.x * 16;
  int n0 = blockIdx.y * 64 + w * 16;
  bool nvalid = (n0 + 16) <= Nc;
  int nb = nvalid ? n0 : (Nc - 16);
  v8f acc = {};
  for (int k0 = 0; k0 < K; k0 += 32) {
    if (k0 + 32 < K) __builtin_prefetch(&A[(size_t)m0 * K + k0 + 32], 0, 1);
    {  // branchless A stage: one float4 per thread, clamp M-tail rows
      int r = tid >> 3, c4 = (tid & 7) * 4;
      int gm = m0 + r;
      float sc = (gm < M) ? 1.f : 0.f;
      int gmc = (gm < M) ? gm : (M - 1);
      float4 f = *(const float4*)(A + (size_t)gmc * K + k0 + c4);
      v4h h4 = {(_Float16)(f.x * sc), (_Float16)(f.y * sc),
                (_Float16)(f.z * sc), (_Float16)(f.w * sc)};
      *(v4h*)(sA + r * 32 + c4) = h4;
    }
    stage_bT(W + (size_t)k0 * Nc + nb, (size_t)Nc, sB[w], lane);
    __syncthreads();
    acc = wmma_f16(frag_a(sA, 32, lane), frag_bT(sB[w], 32, lane), acc);
    __syncthreads();
  }
  if (!nvalid) return;
  int col = nb + (lane & 15);
  float bv = bias[col];
#pragma unroll
  for (int r = 0; r < 8; ++r) {
    int row = m0 + r + 8 * (lane >> 4);
    if (row < M) {
      float v = acc[r] + bv;
      if (ACT) v = silu(v);
      C[(size_t)row * Nc + col] = v;
    }
  }
}

// ---------------------------------------------------------------------------
// Fused pair pipeline. grid = PAIR_M/64 blocks, 128 threads = 4 waves,
// wave w owns row tile m0 = (blockIdx.x*4+w)*16. 20 WMMAs per wave.
// Weights staged transposed (N-major) once per block.
// ---------------------------------------------------------------------------
__global__ void pair_fused(const float* __restrict__ cpos,         // [B,NPAD,3]
                           const unsigned char* __restrict__ mask, // [B,NPAD]
                           const float* __restrict__ centers,      // [32]
                           const float* __restrict__ widthp,       // [1]
                           const float* __restrict__ W1,           // [38,64]
                           const float* __restrict__ b1,           // [64]
                           const float* __restrict__ W2,           // [64,64]
                           const float* __restrict__ b2,           // [64]
                           const float* __restrict__ Wb,           // [3,64,8]
                           float* __restrict__ pair_out,           // [M,64]
                           _Float16* __restrict__ bias_out) {      // [M,32]
  __shared__ __align__(16) _Float16 sW1T[64 * 64];  // [col][k]
  __shared__ __align__(16) _Float16 sW2T[64 * 64];  // [col][k]
  __shared__ __align__(16) _Float16 sWbT[32 * 64];  // [col][k]
  __shared__ __align__(16) _Float16 sF[4][16 * 64], sMd[4][16 * 64],
      sP[4][16 * 64];
  __shared__ float sDist[4][16], sNm[4][16];
  int tid = threadIdx.x, lane = tid & 31, w = tid >> 5;

  for (int t = tid; t < 64 * 64; t += 128) {
    int c = t >> 6, k = t & 63;
    sW1T[t] = (_Float16)(k < 38 ? W1[k * 64 + c] : 0.f);
    sW2T[t] = (_Float16)W2[k * 64 + c];
  }
  for (int t = tid; t < 32 * 64; t += 128) {
    int c = t >> 6, k = t & 63;
    float vv = 0.f;
    if (c < 24) vv = Wb[((c >> 3) * 64 + k) * 8 + (c & 7)];
    sWbT[t] = (_Float16)vv;
  }

  int m0 = (blockIdx.x * 4 + w) * 16;
  if (lane < 16) {
    size_t row = (size_t)m0 + lane;
    int j = (int)(row % NSEQ);
    size_t t2 = row / NSEQ;
    int i = (int)(t2 % NSEQ);
    int b = (int)(t2 / NSEQ);
    float xi = cpos[((size_t)b * NPAD + i) * 3 + 0];
    float yi = cpos[((size_t)b * NPAD + i) * 3 + 1];
    float zi = cpos[((size_t)b * NPAD + i) * 3 + 2];
    float dx = xi - cpos[((size_t)b * NPAD + j) * 3 + 0];
    float dy = yi - cpos[((size_t)b * NPAD + j) * 3 + 1];
    float dz = zi - cpos[((size_t)b * NPAD + j) * 3 + 2];
    float d = sqrtf(dx * dx + dy * dy + dz * dz + 1e-12f);
    float nm = (mask[b * NPAD + i] && mask[b * NPAD + j]) ? 1.f : 0.f;
    sDist[w][lane] = d;
    sNm[w][lane] = nm;
    float ispi = (i < NPROT) ? 1.f : 0.f, ispj = (j < NPROT) ? 1.f : 0.f;
    float pp = ispi * ispj, ll = (1.f - ispi) * (1.f - ispj);
    float pl = 1.f - pp - ll;
    _Float16* f = &sF[w][lane * 64];
    f[32] = (_Float16)(pp * nm);
    f[33] = (_Float16)(pl * nm);
    f[34] = (_Float16)(ll * nm);
    f[35] = (_Float16)(nm * __builtin_amdgcn_rcpf(1e-8f + d));
    f[36] = (_Float16)(fminf(fmaxf(d * 0.05f, 0.f), 1.f) * nm);
    f[37] = (_Float16)nm;
#pragma unroll
    for (int c = 38; c < 64; ++c) f[c] = (_Float16)0.f;
  }
  __syncthreads();

  float wd = fabsf(widthp[0]) + 1e-8f;
  float inv2w2 = __builtin_amdgcn_rcpf(2.f * wd * wd);
  for (int t = lane; t < 512; t += 32) {
    int r = t >> 5, c = t & 31;
    float d = sDist[w][r] - centers[c];
    sF[w][r * 64 + c] = (_Float16)(__expf(-d * d * inv2w2) * sNm[w][r]);
  }
  __syncthreads();

  // MLP1: mid = silu(feats @ W1 + b1)  (K=64 -> 2 steps, 4 N-tiles)
  for (int nt = 0; nt < 4; ++nt) {
    v8f acc = {};
#pragma unroll
    for (int ks = 0; ks < 2; ++ks)
      acc = wmma_f16(frag_a(sF[w] + ks * 32, 64, lane),
                     frag_bT(sW1T + nt * 16 * 64 + ks * 32, 64, lane), acc);
    int col = nt * 16 + (lane & 15);
    float bb = b1[col];
#pragma unroll
    for (int r = 0; r < 8; ++r)
      sMd[w][(r + 8 * (lane >> 4)) * 64 + col] = (_Float16)silu(acc[r] + bb);
  }
  __syncthreads();

  // MLP2: pair = mid @ W2 + b2  -> write f32 pair tensor + keep f16 copy
  for (int nt = 0; nt < 4; ++nt) {
    v8f acc = {};
#pragma unroll
    for (int ks = 0; ks < 2; ++ks)
      acc = wmma_f16(frag_a(sMd[w] + ks * 32, 64, lane),
                     frag_bT(sW2T + nt * 16 * 64 + ks * 32, 64, lane), acc);
    int col = nt * 16 + (lane & 15);
    float bb = b2[col];
#pragma unroll
    for (int r = 0; r < 8; ++r) {
      int row = m0 + r + 8 * (lane >> 4);
      float vv = acc[r] + bb;
      pair_out[(size_t)row * 64 + col] = vv;
      sP[w][(r + 8 * (lane >> 4)) * 64 + col] = (_Float16)vv;
    }
  }
  __syncthreads();

  // per-layer head bias: bias = pair @ Wb_packed  [16x32]
  for (int nt = 0; nt < 2; ++nt) {
    v8f acc = {};
#pragma unroll
    for (int ks = 0; ks < 2; ++ks)
      acc = wmma_f16(frag_a(sP[w] + ks * 32, 64, lane),
                     frag_bT(sWbT + nt * 16 * 64 + ks * 32, 64, lane), acc);
    int col = nt * 16 + (lane & 15);
#pragma unroll
    for (int r = 0; r < 8; ++r) {
      int row = m0 + r + 8 * (lane >> 4);
      bias_out[(size_t)row * 32 + col] = (_Float16)acc[r];
    }
  }
}

// ---------------------------------------------------------------------------
// Attention logits: single WMMA per 16x16 tile (K = DH = 32).
// K^T staging is a straight vectorized copy: per key, DH dims contiguous.
// grid (NPAD/16, NPAD/16, B*H), block 32.
// ---------------------------------------------------------------------------
__global__ void attn_logits(const float* __restrict__ q,
                            const float* __restrict__ k,
                            const _Float16* __restrict__ bias24,
                            const unsigned char* __restrict__ mask,
                            float* __restrict__ logits, int layer) {
  int qt = blockIdx.x * 16, kt = blockIdx.y * 16;
  int bh = blockIdx.z;
  int b = bh >> 3, h = bh & 7;
  int lane = threadIdx.x;
  __shared__ __align__(16) _Float16 sA[16 * 32], sB[16 * 32];
  stage_tile(q + ((size_t)(b * NPAD + qt)) * HID + h * DH, HID, sA, lane, 32);
  stage_tile(k + ((size_t)(b * NPAD + kt)) * HID + h * DH, HID, sB, lane, 32);
  __syncthreads();
  v8f acc = {};
  acc = wmma_f16(frag_a(sA, 32, lane), frag_bT(sB, 32, lane), acc);
  const float scale = 0.17677669529663687f;  // 1/sqrt(32)
  int col = kt + (lane & 15);
  bool colok = (col < NSEQ) && mask[b * NPAD + col];
  int colc = (col < NSEQ) ? col : (NSEQ - 1);
  size_t lbase = (size_t)bh * NPAD * NPAD;
#pragma unroll
  for (int r = 0; r < 8; ++r) {
    int row = qt + r + 8 * (lane >> 4);
    int rowc = (row < NSEQ) ? row : (NSEQ - 1);
    float bia = (float)bias24[(((size_t)b * NSEQ + rowc) * NSEQ + colc) * 32 +
                              layer * 8 + h];
    float v = (colok && row < NSEQ) ? (acc[r] * scale + bia) : -1e9f;
    logits[lbase + (size_t)row * NPAD + col] = v;
  }
}

__global__ void softmax_rows(float* __restrict__ logits) {
  size_t base = ((size_t)blockIdx.y * NPAD + blockIdx.x) * NPAD;
  int lane = threadIdx.x;
  float mx = -3.0e38f;
  for (int c = lane; c < NPAD; c += 32) mx = fmaxf(mx, logits[base + c]);
  for (int o = 16; o > 0; o >>= 1) mx = fmaxf(mx, __shfl_xor(mx, o, 32));
  float sum = 0.f;
  for (int c = lane; c < NPAD; c += 32) {
    float e = __expf(logits[base + c] - mx);
    logits[base + c] = e;
    sum += e;
  }
  for (int o = 16; o > 0; o >>= 1) sum += __shfl_xor(sum, o, 32);
  float inv = __builtin_amdgcn_rcpf(sum);
  for (int c = lane; c < NPAD; c += 32) logits[base + c] *= inv;
}

// attn_out[b,q,h*32+dh..] = probs[bh] @ V ; grid (NPAD/16, 2, B*H), block 32
__global__ void attn_v(const float* __restrict__ probs,
                       const float* __restrict__ v,
                       float* __restrict__ out) {
  int qt = blockIdx.x * 16;
  int dh = blockIdx.y * 16;
  int bh = blockIdx.z;
  int b = bh >> 3, h = bh & 7;
  int lane = threadIdx.x;
  __shared__ __align__(16) _Float16 sA[16 * 32], sB[16 * 32];
  v8f acc = {};
  size_t pbase = ((size_t)bh * NPAD + qt) * NPAD;
  for (int k0 = 0; k0 < NPAD; k0 += 32) {
    stage_tile(probs + pbase + k0, NPAD, sA, lane, 32);
    stage_bT(v + ((size_t)(b * NPAD + k0)) * HID + h * DH + dh, HID, sB, lane);
    __syncthreads();
    acc = wmma_f16(frag_a(sA, 32, lane), frag_bT(sB, 32, lane), acc);
    __syncthreads();
  }
  int col = h * DH + dh + (lane & 15);
#pragma unroll
  for (int r = 0; r < 8; ++r)
    out[((size_t)(b * NPAD + qt + r + 8 * (lane >> 4))) * HID + col] = acc[r];
}

// h = LN(h + y) rowwise over 256; grid = rows, block 32; float4 I/O
__global__ void residual_ln(float* __restrict__ h, const float* __restrict__ y,
                            const float* __restrict__ g,
                            const float* __restrict__ be) {
  size_t base = (size_t)blockIdx.x * HID + threadIdx.x * 8;
  float4 x0 = *(const float4*)&h[base];
  float4 x1 = *(const float4*)&h[base + 4];
  float4 y0 = *(const float4*)&y[base];
  float4 y1 = *(const float4*)&y[base + 4];
  float v[8] = {x0.x + y0.x, x0.y + y0.y, x0.z + y0.z, x0.w + y0.w,
                x1.x + y1.x, x1.y + y1.y, x1.z + y1.z, x1.w + y1.w};
  float s = 0.f;
#pragma unroll
  for (int i = 0; i < 8; ++i) s += v[i];
  for (int o = 16; o > 0; o >>= 1) s += __shfl_xor(s, o, 32);
  float mean = s * (1.f / 256.f);
  float var = 0.f;
#pragma unroll
  for (int i = 0; i < 8; ++i) { float d = v[i] - mean; var += d * d; }
  for (int o = 16; o > 0; o >>= 1) var += __shfl_xor(var, o, 32);
  float inv = rsqrtf(var * (1.f / 256.f) + 1e-5f);
  size_t col = threadIdx.x * 8;
  float4 g0 = *(const float4*)&g[col], g1v = *(const float4*)&g[col + 4];
  float4 b0 = *(const float4*)&be[col], b1v = *(const float4*)&be[col + 4];
  float4 o0 = {(v[0] - mean) * inv * g0.x + b0.x,
               (v[1] - mean) * inv * g0.y + b0.y,
               (v[2] - mean) * inv * g0.z + b0.z,
               (v[3] - mean) * inv * g0.w + b0.w};
  float4 o1 = {(v[4] - mean) * inv * g1v.x + b1v.x,
               (v[5] - mean) * inv * g1v.y + b1v.y,
               (v[6] - mean) * inv * g1v.z + b1v.z,
               (v[7] - mean) * inv * g1v.w + b1v.w};
  *(float4*)&h[base] = o0;
  *(float4*)&h[base + 4] = o1;
}

// ---- small utility kernels -------------------------------------------------
__global__ void fill_f32(float* p, size_t n) {
  size_t i = (size_t)blockIdx.x * blockDim.x + threadIdx.x;
  if (i < n) p[i] = 0.f;
}
__global__ void fill_u8(unsigned char* p, size_t n) {
  size_t i = (size_t)blockIdx.x * blockDim.x + threadIdx.x;
  if (i < n) p[i] = 0;
}
__global__ void add_inplace(float* a, const float* b, size_t n) {
  size_t i = (size_t)blockIdx.x * blockDim.x + threadIdx.x;
  if (i < n) a[i] += b[i];
}
__global__ void compute_starts(const int* pb, int n_p, const int* lb, int n_l,
                               int* starts) {
  if (threadIdx.x == 0 && blockIdx.x == 0) {
    int c[8] = {0, 0, 0, 0, 0, 0, 0, 0};
    for (int i = 0; i < n_p; ++i) c[pb[i]]++;
    for (int i = 0; i < n_l; ++i) c[4 + lb[i]]++;
    int s = 0;
    for (int b = 0; b < 4; ++b) { starts[b] = s; s += c[b]; }
    s = 0;
    for (int b = 0; b < 4; ++b) { starts[4 + b] = s; s += c[4 + b]; }
  }
}
// grid = n nodes, block 256 (= HID)
__global__ void scatter_nodes(const float* __restrict__ emb,
                              const float* __restrict__ pos,
                              const int* __restrict__ bidx,
                              const int* __restrict__ starts, float* h0,
                              float* cpos, unsigned char* mask, int base) {
  int node = blockIdx.x, d = threadIdx.x;
  int b = bidx[node], pin = node - starts[b];
  size_t slot = (size_t)b * NPAD + base + pin;
  h0[slot * HID + d] = emb[(size_t)node * HID + d];
  if (d < 3) cpos[slot * 3 + d] = pos[(size_t)node * 3 + d];
  if (d == 0) mask[slot] = 1;
}
__global__ void gather_nodes(const float* __restrict__ h,
                             const int* __restrict__ bidx,
                             const int* __restrict__ starts,
                             float* __restrict__ out, int base) {
  int node = blockIdx.x, d = threadIdx.x;
  int b = bidx[node], pin = node - starts[b];
  out[(size_t)node * HID + d] = h[((size_t)b * NPAD + base + pin) * HID + d];
}
// grid = B, block 256: out[b, 0..255]=mean, out[b,256..511]=std
__global__ void pool_stats(const float* __restrict__ h,
                           const unsigned char* __restrict__ mask,
                           float* __restrict__ out, int base, int len) {
  int b = blockIdx.x, d = threadIdx.x;
  float s = 0.f, cnt = 0.f;
  for (int r = 0; r < len; ++r)
    if (mask[b * NPAD + base + r]) {
      s += h[((size_t)b * NPAD + base + r) * HID + d];
      cnt += 1.f;
    }
  float c = fmaxf(cnt, 1.f);
  float mean = s / c;
  float var = 0.f;
  for (int r = 0; r < len; ++r)
    if (mask[b * NPAD + base + r]) {
      float df = h[((size_t)b * NPAD + base + r) * HID + d] - mean;
      var += df * df;
    }
  var /= c;
  out[(size_t)b * 512 + d] = mean;
  out[(size_t)b * 512 + 256 + d] = sqrtf(var + 1e-8f);
}

// ---------------------------------------------------------------------------
extern "C" void kernel_launch(void* const* d_in, const int* in_sizes, int n_in,
                              void* d_out, int out_size, void* d_ws,
                              size_t ws_size, hipStream_t stream) {
  (void)n_in; (void)out_size; (void)ws_size;
  // ---- inputs (setup_inputs dict order, params flattened in key order) ----
  const float* protein_feat = (const float*)d_in[0];
  const float* ligand_feat  = (const float*)d_in[1];
  const float* protein_pos  = (const float*)d_in[2];
  const float* ligand_pos   = (const float*)d_in[3];
  const int*   protein_batch= (const int*)d_in[4];
  const int*   ligand_batch = (const int*)d_in[5];
  const float *pW1=(const float*)d_in[6],  *pb1=(const float*)d_in[7];
  const float *pW2=(const float*)d_in[8],  *pb2=(const float*)d_in[9];
  const float *lW1=(const float*)d_in[10], *lb1=(const float*)d_in[11];
  const float *lW2=(const float*)d_in[12], *lb2=(const float*)d_in[13];
  const float *rbf_centers=(const float*)d_in[14];
  const float *rbf_width  =(const float*)d_in[15];
  const float *prW1=(const float*)d_in[16], *prb1=(const float*)d_in[17];
  const float *prW2=(const float*)d_in[18], *prb2=(const float*)d_in[19];
  const float *Wq=(const float*)d_in[20], *bq=(const float*)d_in[21];
  const float *Wk=(const float*)d_in[22], *bk=(const float*)d_in[23];
  const float *Wv=(const float*)d_in[24], *bv=(const float*)d_in[25];
  const float *Wo=(const float*)d_in[26], *bo=(const float*)d_in[27];
  const float *Wb=(const float*)d_in[28];
  const float *g1=(const float*)d_in[29], *be1=(const float*)d_in[30];
  const float *g2=(const float*)d_in[31], *be2=(const float*)d_in[32];
  const float *fW1=(const float*)d_in[33], *fb1=(const float*)d_in[34];
  const float *fW2=(const float*)d_in[35], *fb2=(const float*)d_in[36];

  const int n_p = in_sizes[0] / 64;    // DP = 64
  const int n_l = in_sizes[1] / 224;   // DL = 224

  // ---- workspace allocator (~110 MB total) --------------------------------
  char* wbase = (char*)d_ws;
  size_t off = 0;
  auto alloc = [&](size_t bytes) -> void* {
    void* p = wbase + off;
    off = (off + bytes + 255) & ~(size_t)255;
    return p;
  };
  int*   starts   = (int*)  alloc(8 * sizeof(int));
  float* prot_mid = (float*)alloc((size_t)n_p * HID * 4);
  float* prot_emb = (float*)alloc((size_t)n_p * HID * 4);
  float* lig_mid  = (float*)alloc((size_t)n_l * HID * 4);
  float* lig_emb  = (float*)alloc((size_t)n_l * HID * 4);
  float* h0   = (float*)alloc((size_t)BATCH * NPAD * HID * 4);
  float* h    = (float*)alloc((size_t)BATCH * NPAD * HID * 4);
  float* cpos = (float*)alloc((size_t)BATCH * NPAD * 3 * 4);
  unsigned char* mask = (unsigned char*)alloc((size_t)BATCH * NPAD);
  _Float16* bias24 = (_Float16*)alloc((size_t)PAIR_M * 32 * 2);   // 60 MB
  float* qb = (float*)alloc((size_t)BATCH * NPAD * HID * 4);
  float* kb2= (float*)alloc((size_t)BATCH * NPAD * HID * 4);
  float* vb = (float*)alloc((size_t)BATCH * NPAD * HID * 4);
  float* logits = (float*)alloc((size_t)BATCH * HEADS * NPAD * NPAD * 4); // 33.5 MB
  float* attn_out = (float*)alloc((size_t)BATCH * NPAD * HID * 4);
  float* obuf = (float*)alloc((size_t)BATCH * NPAD * HID * 4);
  float* f1   = (float*)alloc((size_t)BATCH * NPAD * 512 * 4);
  float* f2   = (float*)alloc((size_t)BATCH * NPAD * HID * 4);

  // ---- output regions (return order: prot_out, lig_out, pool_p, pool_l, pair)
  float* out_f = (float*)d_out;
  float* o_prot = out_f;
  float* o_lig  = o_prot + (size_t)n_p * HID;
  float* o_poolp= o_lig  + (size_t)n_l * HID;
  float* o_pooll= o_poolp + (size_t)BATCH * 512;
  float* o_pair = o_pooll + (size_t)BATCH * 512;

  auto gemm = [&](const float* A, const float* W, const float* bias, float* C,
                  int M, int K, int Nc, int act) {
    dim3 g((M + 15) / 16, (Nc + 63) / 64);
    if (act)
      gemm_bias_act<1><<<g, 128, 0, stream>>>(A, W, bias, C, M, K, Nc);
    else
      gemm_bias_act<0><<<g, 128, 0, stream>>>(A, W, bias, C, M, K, Nc);
  };

  // ---- ragged bookkeeping -------------------------------------------------
  compute_starts<<<1, 32, 0, stream>>>(protein_batch, n_p, ligand_batch, n_l,
                                       starts);

  // ---- node embedding MLPs (WMMA GEMMs) -----------------------------------
  gemm(protein_feat, pW1, pb1, prot_mid, n_p, 64, HID, 1);
  gemm(prot_mid, pW2, pb2, prot_emb, n_p, HID, HID, 0);
  gemm(ligand_feat, lW1, lb1, lig_mid, n_l, 224, HID, 1);
  gemm(lig_mid, lW2, lb2, lig_emb, n_l, HID, HID, 0);

  // ---- pad/scatter into [B, NPAD, ...] ------------------------------------
  size_t nh = (size_t)BATCH * NPAD * HID;
  fill_f32<<<(unsigned)((nh + 255) / 256), 256, 0, stream>>>(h0, nh);
  size_t npos = (size_t)BATCH * NPAD * 3;
  fill_f32<<<(unsigned)((npos + 255) / 256), 256, 0, stream>>>(cpos, npos);
  size_t nmsk = (size_t)BATCH * NPAD;
  fill_u8<<<(unsigned)((nmsk + 255) / 256), 256, 0, stream>>>(mask, nmsk);
  scatter_nodes<<<n_p, 256, 0, stream>>>(prot_emb, protein_pos, protein_batch,
                                         starts, h0, cpos, mask, 0);
  scatter_nodes<<<n_l, 256, 0, stream>>>(lig_emb, ligand_pos, ligand_batch,
                                         starts + 4, h0, cpos, mask, NPROT);
  hipMemcpyAsync(h, h0, nh * 4, hipMemcpyDeviceToDevice, stream);

  // ---- fused pair pipeline (feats -> MLP -> pair + head bias) -------------
  pair_fused<<<PAIR_M / 64, 128, 0, stream>>>(cpos, mask, rbf_centers,
                                              rbf_width, prW1, prb1, prW2,
                                              prb2, Wb, o_pair, bias24);

  // ---- transformer layers -------------------------------------------------
  const int Mrows = BATCH * NPAD;  // 2048
  for (int l = 0; l < 3; ++l) {
    gemm(h, Wq + (size_t)l * HID * HID, bq + l * HID, qb, Mrows, HID, HID, 0);
    gemm(h, Wk + (size_t)l * HID * HID, bk + l * HID, kb2, Mrows, HID, HID, 0);
    gemm(h, Wv + (size_t)l * HID * HID, bv + l * HID, vb, Mrows, HID, HID, 0);
    attn_logits<<<dim3(NPAD / 16, NPAD / 16, BATCH * HEADS), 32, 0, stream>>>(
        qb, kb2, bias24, mask, logits, l);
    softmax_rows<<<dim3(NPAD, BATCH * HEADS), 32, 0, stream>>>(logits);
    attn_v<<<dim3(NPAD / 16, 2, BATCH * HEADS), 32, 0, stream>>>(logits, vb,
                                                                 attn_out);
    gemm(attn_out, Wo + (size_t)l * HID * HID, bo + l * HID, obuf, Mrows, HID,
         HID, 0);
    residual_ln<<<Mrows, 32, 0, stream>>>(h, obuf, g1 + l * HID, be1 + l * HID);
    gemm(h, fW1 + (size_t)l * HID * 512, fb1 + l * 512, f1, Mrows, HID, 512, 1);
    gemm(f1, fW2 + (size_t)l * 512 * HID, fb2 + l * HID, f2, Mrows, 512, HID, 0);
    residual_ln<<<Mrows, 32, 0, stream>>>(h, f2, g2 + l * HID, be2 + l * HID);
  }

  // ---- h += h0, gather ragged outputs, masked pooling ---------------------
  add_inplace<<<(unsigned)((nh + 255) / 256), 256, 0, stream>>>(h, h0, nh);
  gather_nodes<<<n_p, 256, 0, stream>>>(h, protein_batch, starts, o_prot, 0);
  gather_nodes<<<n_l, 256, 0, stream>>>(h, ligand_batch, starts + 4, o_lig,
                                        NPROT);
  pool_stats<<<BATCH, 256, 0, stream>>>(h, mask, o_poolp, 0, NPROT);
  pool_stats<<<BATCH, 256, 0, stream>>>(h, mask, o_pooll, NPROT,
                                        NSEQ - NPROT);
}
